// SimpleViT_MoE_79912161509424
// MI455X (gfx1250) — matrix operand order repaired and checked by reading, not compiled, not run on hardware
//
#include <hip/hip_runtime.h>
#include <hip/hip_bf16.h>
#include <math.h>

// ---------------- model constants ----------------
#define BATCH   128
#define SEQ     197            // 196 patches + cls
#define SP      224            // padded seq (7*32) for attention tiling
#define DIM     256
#define NHEAD   8
#define HDIM    32
#define NTOK    (BATCH*SEQ)    // 25216 = 1576*16 exactly
#define MTILES  (NTOK/16)      // 1576
#define KPATCH  768            // 3*16*16
#define FFDIM   1024
#define NEXP    4
#define NCLS    38

typedef __attribute__((ext_vector_type(16))) __bf16 v16bf;
typedef __attribute__((ext_vector_type(8)))  float  v8f;

#define WMMA_BF16(a,b,c) \
  __builtin_amdgcn_wmma_f32_16x16x32_bf16(false,(a),false,(b),(short)0,(c),false,false)

// ---------------- Tensor Data Mover (TDM) support ----------------
#if __has_builtin(__builtin_amdgcn_tensor_load_to_lds)
#define HAVE_TDM 1
#else
#define HAVE_TDM 0
#endif
#if __has_builtin(__builtin_amdgcn_s_wait_tensorcnt)
#define TDM_WAIT() __builtin_amdgcn_s_wait_tensorcnt(0)
#else
#define TDM_WAIT() __builtin_amdgcn_s_waitcnt(0)
#endif

#if HAVE_TDM
typedef __attribute__((ext_vector_type(4))) unsigned int tdm_v4u;
typedef __attribute__((ext_vector_type(8))) int          tdm_v8i;
typedef __attribute__((ext_vector_type(4))) int          tdm_v4i;

// Issue a TDM load of a 2D bf16 tile (tile1 rows x tile0 cols, row stride
// stride0 elements) from global memory into LDS (rows stored compactly).
// D# layout per CDNA5 ISA ch.8: group0 = {count/flags, lds_addr, gaddr};
// group1 = {data_size, tensor dims, tile dims, dim0 stride}; groups 2/3 = 0.
__device__ __forceinline__ void tdm_load_tile_bf16(__bf16* lds_dst, const __bf16* gsrc,
                                                   unsigned tdim0, unsigned tdim1,
                                                   unsigned stride0,
                                                   unsigned tile0, unsigned tile1) {
  unsigned lds_off = (unsigned)(unsigned long long)(void*)lds_dst;       // LDS aperture low 32
  unsigned long long ga = (unsigned long long)(const void*)gsrc;
  tdm_v4u g0;
  g0[0] = 1u;                                                // count=1 valid user descriptor
  g0[1] = lds_off;                                           // lds_addr
  g0[2] = (unsigned)(ga & 0xFFFFFFFFu);                      // global_addr[31:0]
  g0[3] = (unsigned)((ga >> 32) & 0x1FFFFFFu) | (2u << 30);  // addr[56:32] | type=2
  tdm_v8i g1;
  g1[0] = (int)(1u << 16);                                   // data_size=1 (2 bytes)
  g1[1] = (int)((tdim0 & 0xFFFFu) << 16);                    // tensor_dim0[15:0] @ bit48
  g1[2] = (int)(((tdim0 >> 16) & 0xFFFFu) | ((tdim1 & 0xFFFFu) << 16));
  g1[3] = (int)(((tdim1 >> 16) & 0xFFFFu) | ((tile0 & 0xFFFFu) << 16));  // tile_dim0 @ bit112
  g1[4] = (int)(tile1 & 0xFFFFu);                            // tile_dim1 (tile_dim2=0)
  g1[5] = (int)stride0;                                      // tensor_dim0_stride[31:0]
  g1[6] = 0;
  g1[7] = 0;
  tdm_v4i gz = {0, 0, 0, 0};
#if defined(__clang_major__) && (__clang_major__ >= 23)
  tdm_v8i gz8 = {0, 0, 0, 0, 0, 0, 0, 0};
  __builtin_amdgcn_tensor_load_to_lds(g0, g1, gz, gz, gz8, 0);
#else
  __builtin_amdgcn_tensor_load_to_lds(g0, g1, gz, gz, 0);
#endif
}
#endif // HAVE_TDM

// ---- fragment loaders (wave32); contiguous k order, same in A and B ----
// A (16x32 bf16, row-major, stride lda):
//   lanes 0-15 : row=lane,    K = 0..15
//   lanes16-31 : row=lane-16, K = 16..31
__device__ __forceinline__ v16bf load_frag_a(const __bf16* A, int lda) {
  int lane = threadIdx.x & 31;
  int row = lane & 15;
  int kb  = (lane >> 4) * 16;
  const __bf16* p = A + row * lda + kb;
  v16bf f;
#pragma unroll
  for (int i = 0; i < 8; ++i) f[i] = p[i];
#pragma unroll
  for (int i = 0; i < 8; ++i) f[8 + i] = p[8 + i];
  return f;
}

// B (32x16 bf16 viewed as [K][N] row-major, stride ldb):
//   lanes 0-15 : col=lane,    K = 0..15
//   lanes16-31 : col=lane-16, K = 16..31
__device__ __forceinline__ v16bf load_frag_b(const __bf16* B, int ldb) {
  int lane = threadIdx.x & 31;
  int col = lane & 15;
  int kb  = (lane >> 4) * 16;
  v16bf f;
#pragma unroll
  for (int i = 0; i < 16; ++i) f[i] = B[(size_t)(kb + i) * ldb + col];
  return f;
}
// C/D (16x16 f32): vgpr r -> M = r + 8*(lane>=16), N = lane&15

// ---------------- small utility kernels ----------------
__global__ void k_zero_bf16(__bf16* p, int n) {
  int i = blockIdx.x * blockDim.x + threadIdx.x;
  if (i < n) p[i] = (__bf16)0.0f;
}

__global__ void k_cvt_bf16(const float* __restrict__ s, __bf16* __restrict__ d, int n) {
  int i = blockIdx.x * blockDim.x + threadIdx.x;
  if (i < n) d[i] = (__bf16)s[i];
}

// dst[c][r] = src[r][c]  (src is R x C row-major)
__global__ void k_transpose_bf16(const float* __restrict__ src, __bf16* __restrict__ dst,
                                 int R, int C) {
  int i = blockIdx.x * blockDim.x + threadIdx.x;
  if (i < R * C) {
    int r = i / C, c = i % C;
    dst[(size_t)c * R + r] = (__bf16)src[i];
  }
}

__global__ void k_init_cls(float* __restrict__ tok, const float* __restrict__ cls) {
  tok[(size_t)blockIdx.x * SEQ * DIM + threadIdx.x] = cls[threadIdx.x];
}

// ---------------- patch embed: im2col GEMM [16 patches x 768] x [768 x 256] ----------------
__global__ void k_patch_embed(const float* __restrict__ x, const __bf16* __restrict__ Bc,
                              const float* __restrict__ cb, float* __restrict__ tok) {
  __shared__ __attribute__((aligned(16))) __bf16 sA[16 * 32];
  __shared__ __attribute__((aligned(16))) __bf16 sB[32 * 256];
  const int b    = blockIdx.x / 13;
  const int tile = blockIdx.x % 13;
  const int tid = threadIdx.x, wave = tid >> 5, lane = tid & 31;
  const int ncol = wave * 32;
  v8f acc0 = {}, acc1 = {};
  for (int kc = 0; kc < 24; ++kc) {
    for (int i = tid; i < 512; i += 256) {
      int p = i >> 5, kk = i & 31, k = kc * 32 + kk;
      int gp = tile * 16 + p;
      float v = 0.0f;
      if (gp < 196) {
        int py = gp / 14, px = gp % 14;
        int c = k >> 8, rem = k & 255, iy = rem >> 4, ix = rem & 15;
        v = x[(((size_t)b * 3 + c) * 224 + (py * 16 + iy)) * 224 + (px * 16 + ix)];
      }
      sA[i] = (__bf16)v;
    }
#if HAVE_TDM
    if (wave == 0) {
      tdm_load_tile_bf16(sB, Bc + (size_t)kc * 32 * 256, 256, 32, 256, 256, 32);
      TDM_WAIT();
    }
#else
    {
      const uint4* gs = (const uint4*)(Bc + (size_t)kc * 32 * 256);
      uint4* ld = (uint4*)sB;
#pragma unroll
      for (int j = 0; j < 4; ++j) ld[tid + j * 256] = gs[tid + j * 256];
    }
#endif
    if (kc < 23) __builtin_prefetch(Bc + (size_t)(kc + 1) * 32 * 256, 0, 1);
    __syncthreads();
    v16bf a = load_frag_a(sA, 32);
    acc0 = WMMA_BF16(a, load_frag_b(sB + ncol, 256), acc0);
    acc1 = WMMA_BF16(a, load_frag_b(sB + ncol + 16, 256), acc1);
    __syncthreads();
  }
  int n0 = lane & 15, mb = (lane >> 4) * 8;
#pragma unroll
  for (int r = 0; r < 8; ++r) {
    int p = tile * 16 + mb + r;
    if (p < 196) {
      size_t row = ((size_t)b * SEQ + 1 + p) * DIM;
      tok[row + ncol + n0]      = acc0[r] + cb[ncol + n0];
      tok[row + ncol + 16 + n0] = acc1[r] + cb[ncol + 16 + n0];
    }
  }
}

// ---------------- layernorm -> bf16 (1 wave per token) ----------------
__global__ void k_ln(const float* __restrict__ X, const float* __restrict__ g,
                     const float* __restrict__ bt, __bf16* __restrict__ Y) {
  int t = blockIdx.x * 8 + (threadIdx.x >> 5);
  int lane = threadIdx.x & 31;
  const float* xr = X + (size_t)t * DIM + lane * 8;
  float v[8], s = 0.f, s2 = 0.f;
#pragma unroll
  for (int i = 0; i < 8; ++i) { v[i] = xr[i]; s += v[i]; s2 += v[i] * v[i]; }
#pragma unroll
  for (int m = 16; m; m >>= 1) { s += __shfl_xor(s, m, 32); s2 += __shfl_xor(s2, m, 32); }
  float mu = s * (1.0f / DIM);
  float rs = rsqrtf(s2 * (1.0f / DIM) - mu * mu + 1e-5f);
  __bf16* yr = Y + (size_t)t * DIM + lane * 8;
#pragma unroll
  for (int i = 0; i < 8; ++i)
    yr[i] = (__bf16)((v[i] - mu) * rs * g[lane * 8 + i] + bt[lane * 8 + i]);
}

// ---------------- QKV GEMM [NTOK x 256] x [256 x 768], scatter to q/kT/v ----------------
__global__ void k_qkv(const __bf16* __restrict__ A, const __bf16* __restrict__ Bq,
                      const float* __restrict__ bias, __bf16* __restrict__ qb,
                      __bf16* __restrict__ kT, __bf16* __restrict__ vb) {
  __shared__ __attribute__((aligned(16))) __bf16 sA[16 * 32];
  __shared__ __attribute__((aligned(16))) __bf16 sB[32 * 256];
  int mtile = blockIdx.x;
  int nbase = blockIdx.y * 256;
  int tid = threadIdx.x, wave = tid >> 5, lane = tid & 31;
  int ncol = wave * 32;
  v8f acc0 = {}, acc1 = {};
  for (int kc = 0; kc < 8; ++kc) {
#if HAVE_TDM
    if (wave == 0) {
      // A tile: 16 rows x 32 cols, row stride DIM
      tdm_load_tile_bf16(sA, A + ((size_t)mtile * 16) * DIM + kc * 32, 32, 16, DIM, 32, 16);
      // B tile: 32 rows x 256 cols, row stride 768
      tdm_load_tile_bf16(sB, Bq + ((size_t)kc * 32) * 768 + nbase, 256, 32, 768, 256, 32);
      TDM_WAIT();
    }
#else
    for (int i = tid; i < 512; i += 256)
      sA[i] = A[((size_t)mtile * 16 + (i >> 5)) * DIM + kc * 32 + (i & 31)];
    for (int i = tid; i < 8192; i += 256)
      sB[i] = Bq[((size_t)kc * 32 + (i >> 8)) * 768 + nbase + (i & 255)];
#endif
    if (kc < 7) __builtin_prefetch(Bq + ((size_t)(kc + 1) * 32) * 768 + nbase, 0, 1);
    __syncthreads();
    v16bf a = load_frag_a(sA, 32);
    acc0 = WMMA_BF16(a, load_frag_b(sB + ncol, 256), acc0);
    acc1 = WMMA_BF16(a, load_frag_b(sB + ncol + 16, 256), acc1);
    __syncthreads();
  }
  int n0 = lane & 15, mb = (lane >> 4) * 8;
#pragma unroll
  for (int r = 0; r < 8; ++r) {
    int t = mtile * 16 + mb + r;
    int bi = t / SEQ, s = t % SEQ;
#pragma unroll
    for (int half = 0; half < 2; ++half) {
      int n = nbase + ncol + half * 16 + n0;
      float val = (half ? acc1[r] : acc0[r]) + bias[n];
      int which = n >> 8;              // 0=q 1=k 2=v
      int f = n & 31, h = (n >> 5) & 7;
      size_t bh = (size_t)bi * NHEAD + h;
      if (which == 0)      qb[(bh * SP + s) * HDIM + f] = (__bf16)(val * 0.17677669529663689f);
      else if (which == 1) kT[(bh * HDIM + f) * SP + s] = (__bf16)val;
      else                 vb[(bh * SP + s) * HDIM + f] = (__bf16)val;
    }
  }
}

// ---------------- attention: one wave per (b, h, qtile) ----------------
__global__ void k_attn(const __bf16* __restrict__ qb, const __bf16* __restrict__ kT,
                       const __bf16* __restrict__ vb, __bf16* __restrict__ aob) {
  __shared__ float  sc[16 * SP];
  __shared__ __bf16 at[16 * SP];
  int id = blockIdx.x;
  int qt = id % 13; id /= 13;
  int h = id & 7;  int b = id >> 3;
  int lane = threadIdx.x;
  size_t bh = (size_t)b * NHEAD + h;
  const __bf16* qbase = qb + bh * SP * HDIM;
  const __bf16* kbase = kT + bh * HDIM * SP;
  const __bf16* vbase = vb + bh * SP * HDIM;
  int n0 = lane & 15, mb = (lane >> 4) * 8;

  v16bf a = load_frag_a(qbase + (size_t)qt * 16 * HDIM, HDIM);   // K=32 in one frag
  for (int kt = 0; kt < 13; ++kt) {
    v8f acc = {};
    acc = WMMA_BF16(a, load_frag_b(kbase + kt * 16, SP), acc);
#pragma unroll
    for (int r = 0; r < 8; ++r) sc[(mb + r) * SP + kt * 16 + n0] = acc[r];
  }
  for (int kk = 197 + lane; kk < SP; kk += 32)
#pragma unroll
    for (int r = 0; r < 16; ++r) sc[r * SP + kk] = -1e30f;
  __syncthreads();
  if (lane < 16) {
    float* row = sc + lane * SP;
    float mx = -1e30f;
    for (int k = 0; k < SP; ++k) mx = fmaxf(mx, row[k]);
    float sum = 0.f;
    for (int k = 0; k < SP; ++k) { float e = __expf(row[k] - mx); row[k] = e; sum += e; }
    float inv = 1.0f / sum;
    __bf16* ar = at + lane * SP;
    for (int k = 0; k < SP; ++k) ar[k] = (__bf16)(row[k] * inv);
  }
  __syncthreads();
  v8f o0 = {}, o1 = {};
  for (int kc = 0; kc < 7; ++kc) {
    v16bf af = load_frag_a(at + kc * 32, SP);
    o0 = WMMA_BF16(af, load_frag_b(vbase + (size_t)kc * 32 * HDIM, HDIM), o0);
    o1 = WMMA_BF16(af, load_frag_b(vbase + (size_t)kc * 32 * HDIM + 16, HDIM), o1);
  }
#pragma unroll
  for (int r = 0; r < 8; ++r) {
    int q = qt * 16 + mb + r;
    if (q < SEQ) {
      size_t row = ((size_t)b * SEQ + q) * DIM + h * HDIM;
      aob[row + n0]      = (__bf16)o0[r];
      aob[row + 16 + n0] = (__bf16)o1[r];
    }
  }
}

// ---------------- out projection + residual ----------------
__global__ void k_outproj(const __bf16* __restrict__ A, const __bf16* __restrict__ Bo,
                          const float* __restrict__ bias, float* __restrict__ tok) {
  __shared__ __attribute__((aligned(16))) __bf16 sA[16 * 32];
  __shared__ __attribute__((aligned(16))) __bf16 sB[32 * 256];
  int mtile = blockIdx.x;
  int tid = threadIdx.x, wave = tid >> 5, lane = tid & 31;
  int ncol = wave * 32;
  v8f acc0 = {}, acc1 = {};
  for (int kc = 0; kc < 8; ++kc) {
#if HAVE_TDM
    if (wave == 0) {
      tdm_load_tile_bf16(sA, A + ((size_t)mtile * 16) * DIM + kc * 32, 32, 16, DIM, 32, 16);
      tdm_load_tile_bf16(sB, Bo + (size_t)kc * 32 * 256, 256, 32, 256, 256, 32);
      TDM_WAIT();
    }
#else
    for (int i = tid; i < 512; i += 256)
      sA[i] = A[((size_t)mtile * 16 + (i >> 5)) * DIM + kc * 32 + (i & 31)];
    {
      const uint4* gs = (const uint4*)(Bo + (size_t)kc * 32 * 256);
      uint4* ld = (uint4*)sB;
#pragma unroll
      for (int j = 0; j < 4; ++j) ld[tid + j * 256] = gs[tid + j * 256];
    }
#endif
    __syncthreads();
    v16bf a = load_frag_a(sA, 32);
    acc0 = WMMA_BF16(a, load_frag_b(sB + ncol, 256), acc0);
    acc1 = WMMA_BF16(a, load_frag_b(sB + ncol + 16, 256), acc1);
    __syncthreads();
  }
  int n0 = lane & 15, mb = (lane >> 4) * 8;
#pragma unroll
  for (int r = 0; r < 8; ++r) {
    size_t t = (size_t)mtile * 16 + mb + r;
    tok[t * DIM + ncol + n0]      += acc0[r] + bias[ncol + n0];
    tok[t * DIM + ncol + 16 + n0] += acc1[r] + bias[ncol + 16 + n0];
  }
}

// ---------------- LN2 + router top-2 mask ----------------
__global__ void k_ln2_router(const float* __restrict__ X, const float* __restrict__ g,
                             const float* __restrict__ bt, const float* __restrict__ rw,
                             const float* __restrict__ rb, __bf16* __restrict__ Y,
                             float* __restrict__ mask) {
  int t = blockIdx.x * 8 + (threadIdx.x >> 5);
  int lane = threadIdx.x & 31;
  const float* xr = X + (size_t)t * DIM + lane * 8;
  float v[8], s = 0.f, s2 = 0.f;
#pragma unroll
  for (int i = 0; i < 8; ++i) { v[i] = xr[i]; s += v[i]; s2 += v[i] * v[i]; }
#pragma unroll
  for (int m = 16; m; m >>= 1) { s += __shfl_xor(s, m, 32); s2 += __shfl_xor(s2, m, 32); }
  float mu = s * (1.0f / DIM);
  float rs = rsqrtf(s2 * (1.0f / DIM) - mu * mu + 1e-5f);
  float y[8];
  __bf16* yr = Y + (size_t)t * DIM + lane * 8;
#pragma unroll
  for (int i = 0; i < 8; ++i) {
    y[i] = (v[i] - mu) * rs * g[lane * 8 + i] + bt[lane * 8 + i];
    yr[i] = (__bf16)y[i];
  }
  float lg[4];
#pragma unroll
  for (int e = 0; e < 4; ++e) {
    float p = 0.f;
#pragma unroll
    for (int i = 0; i < 8; ++i) p += y[i] * rw[e * DIM + lane * 8 + i];
#pragma unroll
    for (int m = 16; m; m >>= 1) p += __shfl_xor(p, m, 32);
    lg[e] = p;
  }
  if (lane == 0) {
    int i1 = 0;
    for (int e = 1; e < 4; ++e) if (lg[e] + rb[e] > lg[i1] + rb[i1]) i1 = e;
    int i2 = -1;
    for (int e = 0; e < 4; ++e)
      if (e != i1 && (i2 < 0 || lg[e] + rb[e] > lg[i2] + rb[i2])) i2 = e;
    for (int e = 0; e < 4; ++e)
      mask[(size_t)t * 4 + e] = (e == i1 || e == i2) ? 1.0f : 0.0f;
  }
}

// ---------------- fused MoE: 16 tokens per block, experts looped, hidden in LDS ----------------
__global__ void k_moe(const __bf16* __restrict__ X, const __bf16* __restrict__ W1,
                      const float* __restrict__ B1, const __bf16* __restrict__ W2,
                      const float* __restrict__ B2, const float* __restrict__ mask,
                      float* __restrict__ tok) {
  __shared__ __attribute__((aligned(16))) __bf16 sA[16 * 256];
  __shared__ __attribute__((aligned(16))) __bf16 sH[16 * FFDIM];
  __shared__ float sM[16 * 4];
  int mtile = blockIdx.x;
  int tid = threadIdx.x, wave = tid >> 5, lane = tid & 31;
  int n0 = lane & 15, mb = (lane >> 4) * 8;
  if (tid < 64) sM[tid] = mask[(size_t)mtile * 64 + tid];
#if HAVE_TDM
  if (wave == 0) {
    // A block: 16 rows x 256 cols, contiguous
    tdm_load_tile_bf16(sA, X + (size_t)mtile * 16 * DIM, 256, 16, DIM, 256, 16);
    TDM_WAIT();
  }
#else
  for (int i = tid; i < 4096; i += 256)
    sA[i] = X[((size_t)mtile * 16 + (i >> 8)) * DIM + (i & 255)];
#endif
  __syncthreads();
  v8f zero = {};
  v8f out0 = zero, out1 = zero;
  for (int e = 0; e < NEXP; ++e) {
    const __bf16* w1 = W1 + (size_t)e * DIM * FFDIM;
    v8f h[8];
#pragma unroll
    for (int f = 0; f < 8; ++f) h[f] = zero;
    for (int kc = 0; kc < 8; ++kc) {
      v16bf a = load_frag_a(sA + kc * 32, DIM);
#pragma unroll
      for (int f = 0; f < 8; ++f)
        h[f] = WMMA_BF16(a, load_frag_b(w1 + (size_t)kc * 32 * FFDIM + wave * 128 + f * 16, FFDIM), h[f]);
    }
#pragma unroll
    for (int f = 0; f < 8; ++f) {
      int nc = wave * 128 + f * 16 + n0;
      float bias = B1[e * FFDIM + nc];
#pragma unroll
      for (int r = 0; r < 8; ++r) {
        float hv = h[f][r] + bias;
        hv = 0.5f * hv * (1.0f + erff(hv * 0.70710678118654752f));   // exact gelu
        sH[(mb + r) * FFDIM + nc] = (__bf16)hv;
      }
    }
    __syncthreads();
    const __bf16* w2 = W2 + (size_t)e * FFDIM * DIM;
    v8f o0 = zero, o1 = zero;
    for (int kc = 0; kc < 32; ++kc) {
      v16bf a = load_frag_a(sH + kc * 32, FFDIM);
      o0 = WMMA_BF16(a, load_frag_b(w2 + (size_t)kc * 32 * DIM + wave * 32, DIM), o0);
      o1 = WMMA_BF16(a, load_frag_b(w2 + (size_t)kc * 32 * DIM + wave * 32 + 16, DIM), o1);
    }
#pragma unroll
    for (int r = 0; r < 8; ++r) {
      float m = sM[(mb + r) * 4 + e];
      out0[r] += m * (o0[r] + B2[e * DIM + wave * 32 + n0]);
      out1[r] += m * (o1[r] + B2[e * DIM + wave * 32 + 16 + n0]);
    }
    __syncthreads();
  }
#pragma unroll
  for (int r = 0; r < 8; ++r) {
    size_t t = (size_t)mtile * 16 + mb + r;
    tok[t * DIM + wave * 32 + n0]      += out0[r];
    tok[t * DIM + wave * 32 + 16 + n0] += out1[r];
  }
}

// ---------------- classifier head on cls token ----------------
__global__ void k_head(const float* __restrict__ tok, const float* __restrict__ hw,
                       const float* __restrict__ hb, float* __restrict__ out) {
  int b = blockIdx.x, wave = threadIdx.x >> 5, lane = threadIdx.x & 31;
  const float* xr = tok + (size_t)b * SEQ * DIM;
  for (int nc = wave; nc < NCLS; nc += 8) {
    float s = 0.f;
    for (int i = lane; i < DIM; i += 32) s += xr[i] * hw[nc * DIM + i];
#pragma unroll
    for (int m = 16; m; m >>= 1) s += __shfl_xor(s, m, 32);
    if (lane == 0) out[b * NCLS + nc] = s + hb[nc];
  }
}

// ---------------- launcher ----------------
extern "C" void kernel_launch(void* const* d_in, const int* in_sizes, int n_in,
                              void* d_out, int out_size, void* d_ws, size_t ws_size,
                              hipStream_t stream) {
  (void)in_sizes; (void)n_in; (void)out_size; (void)ws_size;
  const float* x      = (const float*)d_in[0];
  const float* conv_w = (const float*)d_in[1];
  const float* conv_b = (const float*)d_in[2];
  const float* cls    = (const float*)d_in[3];
  const float* ln1_g  = (const float*)d_in[4];
  const float* ln1_b  = (const float*)d_in[5];
  const float* qkv_w  = (const float*)d_in[6];
  const float* qkv_b  = (const float*)d_in[7];
  const float* out_w  = (const float*)d_in[8];
  const float* out_b  = (const float*)d_in[9];
  const float* ln2_g  = (const float*)d_in[10];
  const float* ln2_b  = (const float*)d_in[11];
  const float* rw     = (const float*)d_in[12];
  const float* rb     = (const float*)d_in[13];
  const float* e_w1   = (const float*)d_in[14];
  const float* e_b1   = (const float*)d_in[15];
  const float* e_w2   = (const float*)d_in[16];
  const float* e_b2   = (const float*)d_in[17];
  const float* head_w = (const float*)d_in[18];
  const float* head_b = (const float*)d_in[19];
  float* out = (float*)d_out;

  char* ws = (char*)d_ws;
  size_t off = 0;
  auto alloc = [&](size_t bytes) -> char* {
    char* p = ws + off;
    off = (off + bytes + 255) & ~(size_t)255;
    return p;
  };
  const size_t QKV_ELEMS = (size_t)BATCH * NHEAD * SP * HDIM;   // 7,340,032
  float*  tok  = (float*) alloc((size_t)NTOK * DIM * 4);
  __bf16* x2b  = (__bf16*)alloc((size_t)NTOK * DIM * 2);
  __bf16* qb   = (__bf16*)alloc(QKV_ELEMS * 2);
  __bf16* kT   = (__bf16*)alloc(QKV_ELEMS * 2);
  __bf16* vb   = (__bf16*)alloc(QKV_ELEMS * 2);
  __bf16* aob  = (__bf16*)alloc((size_t)NTOK * DIM * 2);
  __bf16* x2c  = (__bf16*)alloc((size_t)NTOK * DIM * 2);
  float*  msk  = (float*) alloc((size_t)NTOK * 4 * 4);
  __bf16* Bc   = (__bf16*)alloc((size_t)KPATCH * DIM * 2);
  __bf16* Bq   = (__bf16*)alloc((size_t)DIM * 768 * 2);
  __bf16* Bo   = (__bf16*)alloc((size_t)DIM * DIM * 2);
  __bf16* W1b  = (__bf16*)alloc((size_t)NEXP * DIM * FFDIM * 2);
  __bf16* W2b  = (__bf16*)alloc((size_t)NEXP * FFDIM * DIM * 2);

  // 0) zero padded q/kT/v
  {
    int n = (int)QKV_ELEMS, g = (n + 255) / 256;
    k_zero_bf16<<<g, 256, 0, stream>>>(qb, n);
    k_zero_bf16<<<g, 256, 0, stream>>>(kT, n);
    k_zero_bf16<<<g, 256, 0, stream>>>(vb, n);
  }
  // 1) weight conversion / transposition to [K][N] bf16
  k_transpose_bf16<<<(256 * 768 + 255) / 256, 256, 0, stream>>>(conv_w, Bc, 256, 768);
  k_transpose_bf16<<<(768 * 256 + 255) / 256, 256, 0, stream>>>(qkv_w, Bq, 768, 256);
  k_transpose_bf16<<<(256 * 256 + 255) / 256, 256, 0, stream>>>(out_w, Bo, 256, 256);
  k_cvt_bf16<<<(NEXP * DIM * FFDIM + 255) / 256, 256, 0, stream>>>(e_w1, W1b, NEXP * DIM * FFDIM);
  k_cvt_bf16<<<(NEXP * FFDIM * DIM + 255) / 256, 256, 0, stream>>>(e_w2, W2b, NEXP * FFDIM * DIM);
  // 2) tokens
  k_init_cls<<<BATCH, DIM, 0, stream>>>(tok, cls);
  k_patch_embed<<<BATCH * 13, 256, 0, stream>>>(x, Bc, conv_b, tok);
  // 3) attention block
  k_ln<<<NTOK / 8, 256, 0, stream>>>(tok, ln1_g, ln1_b, x2b);
  k_qkv<<<dim3(MTILES, 3), 256, 0, stream>>>(x2b, Bq, qkv_b, qb, kT, vb);
  k_attn<<<BATCH * NHEAD * 13, 32, 0, stream>>>(qb, kT, vb, aob);
  k_outproj<<<MTILES, 256, 0, stream>>>(aob, Bo, out_b, tok);
  // 4) MoE block
  k_ln2_router<<<NTOK / 8, 256, 0, stream>>>(tok, ln2_g, ln2_b, rw, rb, x2c, msk);
  k_moe<<<MTILES, 256, 0, stream>>>(x2c, W1b, e_b1, W2b, e_b2, msk, tok);
  // 5) head
  k_head<<<BATCH, 256, 0, stream>>>(tok, head_w, head_b, out);
}
